// LlamaQuantizedMLP_71700184040126
// MI455X (gfx1250) — matrix operand order repaired and checked by reading, not compiled, not run on hardware
//
#include <hip/hip_runtime.h>
#include <cstdint>
#include <cstddef>

// ---------------------------------------------------------------------------
// LlamaQuantizedMLP for MI455X (gfx1250, wave32, WMMA)
//
// Roofline: weights dominate -> minimize bytes.
//   Phase 1: read fp32 weights once (541 MB), per-row abs-max -> scale,
//            write *packed int4* to workspace (67.6 MB).
//   Phase 3/4: GEMMs read 67.6 MB int4, unpack nibble->bf16 (exact: q in
//            [-8,7]), v_wmma_f32_16x16x32_bf16, fold per-channel scale after
//            accumulation. Total ~676 MB -> ~29 us at 23.3 TB/s.
// ---------------------------------------------------------------------------

typedef __attribute__((ext_vector_type(16))) __bf16 v16bf;
typedef __attribute__((ext_vector_type(8)))  __bf16 v8bf;
typedef __attribute__((ext_vector_type(8)))  float  v8f;
typedef __attribute__((ext_vector_type(2)))  int    v2i;

#define HIDDEN 4096
#define INTER  11008
#define NTOK   8      // batch*seq tokens
#define MPAD   16     // WMMA M tile

// workspace layout (bytes); int4-packed weight rows (len/2 bytes per row)
static constexpr size_t OFF_QG = 0;                                        // u8 [INTER][HIDDEN/2]
static constexpr size_t OFF_QU = OFF_QG + (size_t)INTER * HIDDEN / 2;      // u8 [INTER][HIDDEN/2]
static constexpr size_t OFF_QD = OFF_QU + (size_t)INTER * HIDDEN / 2;      // u8 [HIDDEN][INTER/2]
static constexpr size_t OFF_SG = OFF_QD + (size_t)HIDDEN * INTER / 2;      // f32 [INTER]
static constexpr size_t OFF_SU = OFF_SG + sizeof(float) * INTER;           // f32 [INTER]
static constexpr size_t OFF_SD = OFF_SU + sizeof(float) * INTER;           // f32 [HIDDEN]
static constexpr size_t OFF_XB = OFF_SD + sizeof(float) * HIDDEN;          // bf16 [MPAD][HIDDEN]
static constexpr size_t OFF_HB = OFF_XB + 2 * (size_t)MPAD * HIDDEN;       // bf16 [MPAD][INTER]

// ---------------- Phase 1: per-row scale + packed int4 quantize ------------
// one wave32 per weight row; 2*INTER + HIDDEN = 26112 rows total
__global__ __launch_bounds__(256) void quant_rows_kernel(
    const float* __restrict__ wg, const float* __restrict__ wu,
    const float* __restrict__ wd,
    uint8_t* __restrict__ qg, uint8_t* __restrict__ qu, uint8_t* __restrict__ qd,
    float* __restrict__ sg, float* __restrict__ su, float* __restrict__ sd) {
  const int wave = (blockIdx.x * blockDim.x + threadIdx.x) >> 5;
  const int lane = threadIdx.x & 31;

  const float* src;
  uint8_t* q;
  float* s;
  int len, row;
  if (wave < INTER)          { src = wg; q = qg; s = sg; len = HIDDEN; row = wave; }
  else if (wave < 2 * INTER) { src = wu; q = qu; s = su; len = HIDDEN; row = wave - INTER; }
  else                       { src = wd; q = qd; s = sd; len = INTER;  row = wave - 2 * INTER; }

  const float* wrow = src + (size_t)row * len;

  // pass 1: abs-max over the row (wave32, float2-strided)
  const float2* wrow2 = (const float2*)wrow;
  float m = 0.0f;
  for (int j = lane; j < (len >> 1); j += 32) {
    const float2 v = wrow2[j];
    m = fmaxf(m, fmaxf(fabsf(v.x), fabsf(v.y)));
  }
  for (int off = 16; off; off >>= 1) m = fmaxf(m, __shfl_xor(m, off, 32));

  const float scale = fmaxf(m * (1.0f / 7.0f), 1.17549435e-38f);  // finfo.tiny
  const float inv = 1.0f / scale;
  if (lane == 0) s[row] = scale;

  // pass 2: quantize + pack 2x int4 per byte (row is cache-resident)
  uint8_t* qrow = q + (size_t)row * (len >> 1);
  for (int j = lane; j < (len >> 1); j += 32) {
    const float2 v = wrow2[j];
    float q0 = fminf(fmaxf(rintf(v.x * inv), -8.0f), 7.0f);  // RNE = jnp.round
    float q1 = fminf(fmaxf(rintf(v.y * inv), -8.0f), 7.0f);
    qrow[j] = (uint8_t)(((int)q0 & 0xF) | (((int)q1 & 0xF) << 4));
  }
}

// ---------------- Phase 2: x -> bf16, pad M to 16 --------------------------
__global__ __launch_bounds__(256) void pack_x_kernel(const float* __restrict__ x,
                                                     __bf16* __restrict__ xb) {
  const int idx = blockIdx.x * blockDim.x + threadIdx.x;
  if (idx >= MPAD * HIDDEN) return;
  const int mrow = idx >> 12;  // /HIDDEN
  const float v = (mrow < NTOK) ? x[idx] : 0.0f;
  xb[idx] = (__bf16)v;
}

// --------------- WMMA fragment helpers (contiguous-K order, same for A and B) -----------
// A 16x32 bf16: per lane (M = lane&15, hs = lane>>4):
//   VGPR0..3 <- K = hs*16 + [0..7],  VGPR4..7 <- K = hs*16 + 8 + [0..7]
__device__ __forceinline__ v16bf load_a_frag(const __bf16* __restrict__ row,
                                             int k, int hs) {
  const v8bf c0 = *(const v8bf*)(row + k + hs * 16);
  const v8bf c1 = *(const v8bf*)(row + k + 8 + hs * 16);
  v16bf a;
#pragma unroll
  for (int j = 0; j < 8; ++j) { a[j] = c0[j]; a[8 + j] = c1[j]; }
  return a;
}

// B 32x16 bf16: per lane (N = lane&15): contiguous K chunk of 16 starting at
// k + (lane>>4)*16. Source: 8 bytes of packed int4 (nibble j = weight j,
// two's complement, exact in bf16). One v_bfe_i32 + cvt per weight.
__device__ __forceinline__ v16bf cvt16_i4_bf16(v2i raw) {
  v16bf r;
#pragma unroll
  for (int w = 0; w < 2; ++w) {
    const int word = raw[w];
#pragma unroll
    for (int nib = 0; nib < 8; ++nib) {
      r[8 * w + nib] = (__bf16)(float)((word << (28 - 4 * nib)) >> 28);
    }
  }
  return r;
}

// ---------------- Phase 3: gate/up WMMA + SwiGLU ---------------------------
// One block per 16-channel tile; 8 waves split K=4096 (512 each), LDS reduce.
__global__ __launch_bounds__(256) void gate_up_swiglu_kernel(
    const __bf16* __restrict__ xb,
    const uint8_t* __restrict__ qg, const uint8_t* __restrict__ qu,
    const float* __restrict__ sg, const float* __restrict__ su,
    __bf16* __restrict__ hb) {
  __shared__ float redg[8][256];
  __shared__ float redu[8][256];

  const int i0 = blockIdx.x * 16;
  const int wave = threadIdx.x >> 5;
  const int lane = threadIdx.x & 31;
  const int hs = lane >> 4;
  const int mrow = lane & 15;
  const int kbeg = wave * (HIDDEN / 8);  // 512 K per wave

  v8f accg = {};
  v8f accu = {};
  const __bf16* arow = xb + (size_t)mrow * HIDDEN;
  const uint8_t* grow = qg + (size_t)(i0 + mrow) * (HIDDEN / 2);
  const uint8_t* urow = qu + (size_t)(i0 + mrow) * (HIDDEN / 2);

#pragma unroll 2
  for (int k = kbeg; k < kbeg + HIDDEN / 8; k += 32) {
    __builtin_prefetch(grow + (k >> 1) + 128, 0, 1);  // -> global_prefetch_b8
    __builtin_prefetch(urow + (k >> 1) + 128, 0, 1);
    const v16bf a = load_a_frag(arow, k, hs);
    const v16bf bg = cvt16_i4_bf16(*(const v2i*)(grow + (k >> 1) + hs * 8));
    const v16bf bu = cvt16_i4_bf16(*(const v2i*)(urow + (k >> 1) + hs * 8));
    accg = __builtin_amdgcn_wmma_f32_16x16x32_bf16(false, a, false, bg,
                                                   (short)0, accg, false, false);
    accu = __builtin_amdgcn_wmma_f32_16x16x32_bf16(false, a, false, bu,
                                                   (short)0, accu, false, false);
  }

#pragma unroll
  for (int r = 0; r < 8; ++r) {
    redg[wave][lane * 8 + r] = accg[r];
    redu[wave][lane * 8 + r] = accu[r];
  }
  __syncthreads();

  // thread t owns C element (lane l = t>>3, acc reg r = t&7)
  const int l = threadIdx.x >> 3;
  const int r = threadIdx.x & 7;
  float g = 0.0f, u = 0.0f;
#pragma unroll
  for (int w = 0; w < 8; ++w) { g += redg[w][l * 8 + r]; u += redu[w][l * 8 + r]; }

  const int n = i0 + (l & 15);
  g *= sg[n];
  u *= su[n];
  const float hval = __fdividef(g, 1.0f + __expf(-g)) * u;  // silu(gate)*up
  const int mm = r + ((l >> 4) << 3);  // C layout: M = r + 8*(lane>=16)
  hb[(size_t)mm * INTER + n] = (__bf16)hval;
}

// ---------------- Phase 4: down projection WMMA ----------------------------
__global__ __launch_bounds__(256) void down_proj_kernel(
    const __bf16* __restrict__ hb, const uint8_t* __restrict__ qd,
    const float* __restrict__ sd, float* __restrict__ out) {
  __shared__ float red[8][256];

  const int i0 = blockIdx.x * 16;
  const int wave = threadIdx.x >> 5;
  const int lane = threadIdx.x & 31;
  const int hs = lane >> 4;
  const int mrow = lane & 15;
  const int kbeg = wave * (INTER / 8);  // 1376 = 43*32 K per wave

  v8f acc = {};
  const __bf16* arow = hb + (size_t)mrow * INTER;
  const uint8_t* drow = qd + (size_t)(i0 + mrow) * (INTER / 2);

#pragma unroll 2
  for (int k = kbeg; k < kbeg + INTER / 8; k += 32) {
    __builtin_prefetch(drow + (k >> 1) + 128, 0, 1);
    const v16bf a = load_a_frag(arow, k, hs);
    const v16bf b = cvt16_i4_bf16(*(const v2i*)(drow + (k >> 1) + hs * 8));
    acc = __builtin_amdgcn_wmma_f32_16x16x32_bf16(false, a, false, b,
                                                  (short)0, acc, false, false);
  }

#pragma unroll
  for (int r = 0; r < 8; ++r) red[wave][lane * 8 + r] = acc[r];
  __syncthreads();

  const int l = threadIdx.x >> 3;
  const int r = threadIdx.x & 7;
  float d = 0.0f;
#pragma unroll
  for (int w = 0; w < 8; ++w) d += red[w][l * 8 + r];

  const int n = i0 + (l & 15);
  const int mm = r + ((l >> 4) << 3);
  if (mm < NTOK) out[(size_t)mm * HIDDEN + n] = d * sd[n];
}

// ---------------------------------------------------------------------------
extern "C" void kernel_launch(void* const* d_in, const int* in_sizes, int n_in,
                              void* d_out, int out_size, void* d_ws, size_t ws_size,
                              hipStream_t stream) {
  (void)in_sizes; (void)n_in; (void)out_size; (void)ws_size;
  const float* x  = (const float*)d_in[0];
  const float* wg = (const float*)d_in[1];
  const float* wu = (const float*)d_in[2];
  const float* wd = (const float*)d_in[3];

  char* ws = (char*)d_ws;
  uint8_t* qg = (uint8_t*)(ws + OFF_QG);
  uint8_t* qu = (uint8_t*)(ws + OFF_QU);
  uint8_t* qd = (uint8_t*)(ws + OFF_QD);
  float*   sg = (float*)(ws + OFF_SG);
  float*   su = (float*)(ws + OFF_SU);
  float*   sd = (float*)(ws + OFF_SD);
  __bf16*  xb = (__bf16*)(ws + OFF_XB);
  __bf16*  hb = (__bf16*)(ws + OFF_HB);
  float*  out = (float*)d_out;

  // 26112 rows, 8 waves per block
  quant_rows_kernel<<<(2 * INTER + HIDDEN) / 8, 256, 0, stream>>>(
      wg, wu, wd, qg, qu, qd, sg, su, sd);
  pack_x_kernel<<<(MPAD * HIDDEN + 255) / 256, 256, 0, stream>>>(x, xb);
  gate_up_swiglu_kernel<<<INTER / 16, 256, 0, stream>>>(xb, qg, qu, sg, su, hb);
  down_proj_kernel<<<HIDDEN / 16, 256, 0, stream>>>(hb, qd, sd, out);
}
